// LinearBlockEncoder_73478300500639
// MI455X (gfx1250) — 8-bit matrix operand order repaired and checked by reading, compiled, not run on hardware
//
#include <hip/hip_runtime.h>


typedef __attribute__((ext_vector_type(8))) int v8i;
typedef __attribute__((ext_vector_type(4))) int v4i;

#define K_MSG 1024
#define N_CODE 2048
#define NKG 16

#if defined(__gfx1250__) &&                                          \
    __has_builtin(__builtin_amdgcn_global_load_async_to_lds_b128) && \
    __has_builtin(__builtin_amdgcn_s_wait_asynccnt)
#define USE_ASYNC_LDS 1
typedef __attribute__((address_space(1))) v4i* gptr_v4i;
typedef __attribute__((address_space(3))) v4i* lptr_v4i;
#else
#define USE_ASYNC_LDS 0
#endif

__global__ __launch_bounds__(256) void pack_a(const int* __restrict__ b,
                                              uint4* __restrict__ Ap) {
  const int chunk = blockIdx.x * 8 + (threadIdx.x >> 5);
  const int lane  = threadIdx.x & 31;
  const int mtile = chunk >> 4;
  const int kg    = chunk & (NKG - 1);
  const int half  = lane >> 4;
  const int row   = mtile * 16 + (lane & 15);

  unsigned int u[8] __attribute__((aligned(16)));
#pragma unroll
  for (int v = 0; v < 8; ++v) {
    const int kb = (v >> 1) * 16 + half * 8 + (v & 1) * 4;
    const int4 w = *(const int4*)(b + (size_t)row * K_MSG + kg * 64 + kb);
    u[v] = (unsigned)(w.x & 1) | ((unsigned)(w.y & 1) << 8) |
           ((unsigned)(w.z & 1) << 16) | ((unsigned)(w.w & 1) << 24);
  }
  uint4* dst = Ap + (size_t)chunk * 64 + lane * 2;
  dst[0] = *(const uint4*)&u[0];
  dst[1] = *(const uint4*)&u[4];
}

__global__ __launch_bounds__(256) void pack_b(const int* __restrict__ G,
                                              uint4* __restrict__ Bp) {
  const int chunk = blockIdx.x * 8 + (threadIdx.x >> 5);
  const int lane  = threadIdx.x & 31;
  const int ntile = chunk >> 4;
  const int kg    = chunk & (NKG - 1);
  const int half  = lane >> 4;
  const int n     = ntile * 16 + (lane & 15);

  unsigned int u[8] __attribute__((aligned(16)));
#pragma unroll
  for (int v = 0; v < 8; ++v) {
    const int kb = (v >> 1) * 16 + half * 8 + (v & 1) * 4;
    const size_t base = (size_t)(kg * 64 + kb) * N_CODE + n;
    u[v] = (unsigned)(G[base] & 1) |
           ((unsigned)(G[base + N_CODE] & 1) << 8) |
           ((unsigned)(G[base + 2 * N_CODE] & 1) << 16) |
           ((unsigned)(G[base + 3 * N_CODE] & 1) << 24);
  }
  uint4* dst = Bp + (size_t)chunk * 64 + lane * 2;
  dst[0] = *(const uint4*)&u[0];
  dst[1] = *(const uint4*)&u[4];
}

__global__ __launch_bounds__(256) void gf2_gemm_frag(
    const uint4* __restrict__ Ap, const uint4* __restrict__ Bp,
    float* __restrict__ out) {
  __shared__ uint4 ldsB[2][8 * 64];

  const int t    = threadIdx.x;
  const int wave = t >> 5;
  const int lane = t & 31;
  const int half = lane >> 4;
  const int l16  = lane & 15;

  const int mtile0 = blockIdx.y * 16 + wave * 2;
  const int nt0    = blockIdx.x * 8;

  const int snt = t >> 5;
  const int sl  = t & 31;

  v8i acc[2][8] = {};

  auto stageB = [&](int buf, int kg) {
    const uint4* src = Bp + (size_t)((nt0 + snt) * 16 + kg) * 64 + sl * 2;
    uint4* dst = &ldsB[buf][snt * 64 + sl * 2];
#if USE_ASYNC_LDS
    __builtin_amdgcn_global_load_async_to_lds_b128(
        (gptr_v4i)(uintptr_t)src, (lptr_v4i)(uintptr_t)dst, 0, 0);
    __builtin_amdgcn_global_load_async_to_lds_b128(
        (gptr_v4i)(uintptr_t)src, (lptr_v4i)(uintptr_t)dst, 16, 0);
#else
    dst[0] = src[0];
    dst[1] = src[1];
#endif
  };

  stageB(0, 0);

  for (int kg = 0; kg < NKG; ++kg) {
    const int cur = kg & 1;
#if USE_ASYNC_LDS
    __builtin_amdgcn_s_wait_asynccnt(0);
#endif
    __syncthreads();
    if (kg + 1 < NKG) stageB(cur ^ 1, kg + 1);

    v8i af[2];
#pragma unroll
    for (int mt = 0; mt < 2; ++mt) {
      const uint4* asrc =
          Ap + (size_t)((mtile0 + mt) * 16 + kg) * 64 + lane * 2;
      ((uint4*)&af[mt])[0] = asrc[0];
      ((uint4*)&af[mt])[1] = asrc[1];
    }

#pragma unroll
    for (int nt = 0; nt < 8; ++nt) {
      v8i bfrag;
      ((uint4*)&bfrag)[0] = ldsB[cur][nt * 64 + lane * 2 + 0];
      ((uint4*)&bfrag)[1] = ldsB[cur][nt * 64 + lane * 2 + 1];
      acc[0][nt] = __builtin_amdgcn_wmma_i32_16x16x64_iu8(
          false, af[0], false, bfrag, acc[0][nt], false, false);
      acc[1][nt] = __builtin_amdgcn_wmma_i32_16x16x64_iu8(
          false, af[1], false, bfrag, acc[1][nt], false, false);
    }
  }

#pragma unroll
  for (int mt = 0; mt < 2; ++mt) {
    const int row0 = (mtile0 + mt) * 16 + half * 8;
#pragma unroll
    for (int nt = 0; nt < 8; ++nt) {
#pragma unroll
      for (int r = 0; r < 8; ++r) {
        out[(size_t)(row0 + r) * N_CODE + (nt0 + nt) * 16 + l16] =
            (float)(acc[mt][nt][r] & 1);
      }
    }
  }
}

extern "C" void kernel_launch(void* const* d_in, const int* in_sizes, int n_in,
                              void* d_out, int out_size, void* d_ws,
                              size_t ws_size, hipStream_t stream) {
  const int* b = (const int*)d_in[0];
  const int* G = (const int*)d_in[1];
  float* out   = (float*)d_out;

  const int M      = in_sizes[0] / K_MSG;
  const int mtiles = M / 16;
  const int ntiles = N_CODE / 16;

  uint4* Ap = (uint4*)d_ws;
  uint4* Bp = (uint4*)((char*)d_ws + (size_t)M * K_MSG);

  pack_a<<<dim3(mtiles * NKG / 8), 256, 0, stream>>>(b, Ap);
  pack_b<<<dim3(ntiles * NKG / 8), 256, 0, stream>>>(G, Bp);

  dim3 grid(ntiles / 8, M / 256);
  gf2_gemm_frag<<<grid, 256, 0, stream>>>(Ap, Bp, out);
}
